// RNN_f3_34239479284342
// MI455X (gfx1250) — matrix operand order repaired and checked by reading, not compiled, not run on hardware
//
#include <hip/hip_runtime.h>
#include <stdint.h>

// MI455X / gfx1250: wave32, WMMA 16x16x32 bf16, 320KB LDS per WGP.
//
// Key algebraic fact: reference output y = out[:, -1, :] @ W2.T + b2 depends
// ONLY on batch element 255 (RNN recurrence is per-batch independent). So we
// run a single-sequence RNN: 512 sequential steps of h = relu(pre_t + W_hh h),
// H = 512. Latency-bound -> one persistent workgroup (32 waves, one WGP),
// W_hh resident in bf16 split across VGPRs (8 frags/wave) + LDS (256KB),
// per-step sync = one workgroup barrier, matvec via v_wmma_f32_16x16x32_bf16.

typedef __attribute__((ext_vector_type(16))) __bf16 v16bf;
typedef __attribute__((ext_vector_type(8)))  float  v8f;

#define S_ 512
#define B_ 256
#define I_ 64
#define H_ 512
#define O_ 25

__device__ __forceinline__ unsigned short f32_to_bf16(float f) {
    union { float f; uint32_t u; } v; v.f = f;
    uint32_t u = v.u;
    return (unsigned short)((u + 0x7fffu + ((u >> 16) & 1u)) >> 16);
}

// ---------------------------------------------------------------------------
// pre[t][j] = dot(x[t, 255, :], W_ih[j, :]) + b_ih[j] + b_hh[j]   (fp32)
// Tiny (33 MFLOP) -> plain VALU kernel, x row broadcast through LDS.
// ---------------------------------------------------------------------------
__global__ void rnn_prep_pre(const float* __restrict__ x,
                             const float* __restrict__ Wih,
                             const float* __restrict__ bih,
                             const float* __restrict__ bhh,
                             float* __restrict__ pre) {
    __shared__ float xr[I_];
    const int t = blockIdx.x;
    const int j = threadIdx.x;
    if (j < I_) xr[j] = x[(t * B_ + (B_ - 1)) * I_ + j];
    __syncthreads();
    float s = bih[j] + bhh[j];
    const float* w = Wih + j * I_;
#pragma unroll 8
    for (int i = 0; i < I_; ++i) s += xr[i] * w[i];
    pre[t * H_ + j] = s;
}

// ---------------------------------------------------------------------------
// Swizzle W_hh (fp32 row-major, A[row][k] = W_hh[row*512+k]) into bf16 WMMA
// A-fragments for 16x16x32: fragment f = rowTile*16 + kTile, 1KB each,
// per-lane 16 bf16 values in VGPR order (ISA 16-bit A layout):
//   row = rowTile*16 + (lane&15)
//   k   = kTile*32 + (v>=4 ? 16:0) + (lane>=16 ? 8:0) + 2*(v&3) + p
// ---------------------------------------------------------------------------
__global__ void rnn_prep_whh(const float* __restrict__ Whh,
                             unsigned short* __restrict__ wf) {
    const int f = blockIdx.x;       // 0..511
    const int lane = threadIdx.x;   // 0..31
    const int rowTile = f >> 4, kt = f & 15;
    const int row = rowTile * 16 + (lane & 15);
    unsigned short* o = wf + f * 512 + lane * 16;
#pragma unroll
    for (int j = 0; j < 16; ++j) {
        const int v = j >> 1, p = j & 1;
        const int k = kt * 32 + ((v & 4) ? 16 : 0) + ((lane & 16) ? 8 : 0)
                    + 2 * (v & 3) + p;
        o[j] = f32_to_bf16(Whh[row * H_ + k]);
    }
}

// ---------------------------------------------------------------------------
// Persistent scan: one workgroup, 32 waves. Wave w owns rows [16w, 16w+16).
// A-fragments: kTiles 0..7 in LDS (256KB total), kTiles 8..15 in VGPRs.
// B = current h (bf16) replicated over all 16 columns; per-lane B fragment is
// two 8-element K-slices of h (k = 8*half.., 16+8*half..), same k order as A.
// C layout: lanes 0/16 (N==0) hold the 16 row results -> they apply pre+relu,
// write bf16 h to the double-buffered LDS h array and fp32 h to Hout.
// ---------------------------------------------------------------------------
__global__ void __launch_bounds__(1024, 1)
rnn_scan(const unsigned short* __restrict__ wf,  // swizzled W_hh frags
         const float* __restrict__ pre,          // S x H fp32
         float* __restrict__ Hout) {             // S x H fp32
    extern __shared__ unsigned char smem[];
    unsigned short* wlds = (unsigned short*)smem;              // 256 KB
    unsigned short* hbuf = (unsigned short*)(smem + 262144);   // 2 x 512 bf16

    const int tid  = threadIdx.x;
    const int lane = tid & 31;
    const int w    = tid >> 5;             // wave id == rowTile

    // Stage LDS-resident A fragments (kTiles 0..7): 32B per lane per frag.
#pragma unroll
    for (int kt = 0; kt < 8; ++kt) {
        const int g = w * 16 + kt;
        const float4* src = (const float4*)(wf + g * 512 + lane * 16);
        float4* dst = (float4*)(wlds + (w * 8 + kt) * 512 + lane * 16);
        dst[0] = src[0];
        dst[1] = src[1];
    }
    // Register-resident A fragments (kTiles 8..15): 64 VGPRs/wave.
    v16bf areg[8];
#pragma unroll
    for (int kt = 0; kt < 8; ++kt) {
        const int g = w * 16 + 8 + kt;
        areg[kt] = *(const v16bf*)(wf + g * 512 + lane * 16);
    }
    // h0 = 0 in both buffers.
    if (tid < H_) { hbuf[tid] = 0; hbuf[H_ + tid] = 0; }
    __syncthreads();

    const int  half    = (lane >> 4) & 1;        // K half / row half select
    const bool writer  = (lane & 15) == 0;       // lanes holding column N==0
    const int  rowBase = w * 16 + half * 8;

    for (int t = 0; t < S_; ++t) {
        const unsigned short* hc = hbuf + (t & 1) * H_;
        unsigned short*       hn = hbuf + ((t + 1) & 1) * H_;

        v8f acc0 = {}; v8f acc1 = {};
#pragma unroll
        for (int kt = 0; kt < 16; ++kt) {
            v16bf b;
#pragma unroll
            for (int s = 0; s < 16; ++s)
                b[s] = ((const __bf16*)hc)[kt * 32 + ((s & 8) ? 16 : 0) + half * 8 + (s & 7)];
            const v16bf a = (kt < 8)
                ? *(const v16bf*)(wlds + (w * 8 + kt) * 512 + lane * 16)
                : areg[kt - 8];
            if (kt & 1)
                acc1 = __builtin_amdgcn_wmma_f32_16x16x32_bf16(
                           false, a, false, b, (short)0, acc1, false, false);
            else
                acc0 = __builtin_amdgcn_wmma_f32_16x16x32_bf16(
                           false, a, false, b, (short)0, acc0, false, false);
        }

        if (writer) {
            const float* p  = pre  + t * H_ + rowBase;
            float*       ho = Hout + t * H_ + rowBase;
#pragma unroll
            for (int r = 0; r < 8; ++r) {
                float vv = acc0[r] + acc1[r] + p[r];
                vv = fmaxf(vv, 0.0f);
                ho[r] = vv;
                hn[rowBase + r] = f32_to_bf16(vv);
            }
        }
        __syncthreads();   // one barrier per step: next-t reads hn
    }
}

// ---------------------------------------------------------------------------
// y[t][o] = dot(H[t], W2[o]) + b2[o].  6.5 MFLOP -> wave-per-output VALU.
// Block = 25 waves (800 threads), grid = 512 (one block per t).
// ---------------------------------------------------------------------------
__global__ void rnn_proj(const float* __restrict__ Hout,
                         const float* __restrict__ W2,
                         const float* __restrict__ b2,
                         float* __restrict__ y) {
    const int t = blockIdx.x;
    const int o = threadIdx.x >> 5;   // 0..24
    const int l = threadIdx.x & 31;
    const float* h  = Hout + t * H_;
    const float* wv = W2 + o * H_;
    float s = 0.f;
    for (int k = l; k < H_; k += 32) s += h[k] * wv[k];
#pragma unroll
    for (int off = 16; off; off >>= 1) s += __shfl_down(s, off, 32);
    if (l == 0) y[t * O_ + o] = s + b2[o];
}

extern "C" void kernel_launch(void* const* d_in, const int* in_sizes, int n_in,
                              void* d_out, int out_size, void* d_ws, size_t ws_size,
                              hipStream_t stream) {
    const float* x   = (const float*)d_in[0];
    const float* Wih = (const float*)d_in[1];
    const float* Whh = (const float*)d_in[2];
    const float* bih = (const float*)d_in[3];
    const float* bhh = (const float*)d_in[4];
    const float* W2  = (const float*)d_in[5];
    const float* b2  = (const float*)d_in[6];
    float* y = (float*)d_out;

    uint8_t* ws = (uint8_t*)d_ws;
    float*          pre  = (float*)ws;                                   // 1 MB
    unsigned short* wf   = (unsigned short*)(ws + (1u << 20));           // 512 KB
    float*          Hbuf = (float*)(ws + (1u << 20) + (1u << 19));       // 1 MB

    rnn_prep_pre<<<S_, H_, 0, stream>>>(x, Wih, bih, bhh, pre);
    rnn_prep_whh<<<512, 32, 0, stream>>>(Whh, wf);
    // dynamic LDS: 256KB weight fragments + 2KB h double buffer
    rnn_scan<<<1, 1024, 262144 + 2048, stream>>>(wf, pre, Hbuf);
    rnn_proj<<<S_, 800, 0, stream>>>(Hbuf, W2, b2, y);
}
